// LinkPredictorBase_79096117723241
// MI455X (gfx1250) — matrix operand order repaired and checked by reading, not compiled, not run on hardware
//
#include <hip/hip_runtime.h>
#include <hip/hip_bf16.h>
#include <stdint.h>

typedef __attribute__((ext_vector_type(16))) __bf16 v16bf;
typedef __attribute__((ext_vector_type(8)))  __bf16 v8bf;
typedef __attribute__((ext_vector_type(8)))  float  v8f;
typedef __attribute__((ext_vector_type(4)))  float  v4f;

#define IN_CH   128
#define KDIM    256        // 2*IN_CH
#define HID     256
#define HIDP    257        // +1 f32 pad -> conflict-free layer-2 reads
#define KSTEPS  8          // KDIM / 32
#define ZROW    33         // v8bf per half-row: 264 bf16 (256 + 16B pad)
#define MWG     64         // edges per workgroup
#define MTILES  4          // MWG / 16
#define WFRAG   65536      // 16 nt * 8 ks * 32 lanes * 16 elems

__device__ __forceinline__ __bf16 bf16_hi(float f) {
    unsigned u = __builtin_bit_cast(unsigned, f);
    unsigned short h = (unsigned short)(u >> 16);
    return __builtin_bit_cast(__bf16, h);
}
__device__ __forceinline__ float bf16_to_f(__bf16 b) {
    unsigned u = ((unsigned)__builtin_bit_cast(unsigned short, b)) << 16;
    return __builtin_bit_cast(float, u);
}
__device__ __forceinline__ v16bf cat8(v8bf a, v8bf b) {
    return __builtin_shufflevector(a, b, 0,1,2,3,4,5,6,7,8,9,10,11,12,13,14,15);
}

// ---------------------------------------------------------------------------
// Repack W1 (fp32 [K=256][N=256]) into per-lane WMMA B-fragments, bf16 hi/lo.
// ws: [hi: WFRAG][lo: WFRAG].  idx = ((nt*8+ks)*32+lane)*16+i
//   n = nt*16 + (lane&15);  k = ks*32 + (i>>3)*16 + (lane>>4)*8 + (i&7)
// ---------------------------------------------------------------------------
__global__ void repack_w1(const float* __restrict__ W1, __bf16* __restrict__ ws) {
    int tid  = blockIdx.x * blockDim.x + threadIdx.x;   // 0..65535
    int i    = tid & 15;
    int lane = (tid >> 4) & 31;
    int ks   = (tid >> 9) & 7;
    int nt   = tid >> 12;
    int n    = nt * 16 + (lane & 15);
    int k    = ks * 32 + (i >> 3) * 16 + (lane >> 4) * 8 + (i & 7);
    float w  = W1[k * HID + n];
    __bf16 hi = bf16_hi(w);
    __bf16 lo = bf16_hi(w - bf16_to_f(hi));
    ws[tid]         = hi;
    ws[WFRAG + tid] = lo;
}

// ---------------------------------------------------------------------------
// LDS: gather tile as v8bf (16B-aligned -> ds_load_b128), hi/lo interleaved
// per row so every A-fragment address = one lane base + 16-bit const offset.
// Row block: z[row][0][0..32] = hi half-row, z[row][1][0..32] = lo half-row.
// ---------------------------------------------------------------------------
union SMem {
    struct { v8bf  z[MWG][2][ZROW]; } g;                 // 64*2*33*16 = 67584 B
    struct { float h[MWG][HIDP]; float red[MWG][4]; } r; // 66816 B
};

__global__ __launch_bounds__(256)
void edge_mlp(const long long* __restrict__ ei, const float* __restrict__ x,
              const __bf16* __restrict__ wsB,
              const float* __restrict__ b1, const float* __restrict__ W2,
              const float* __restrict__ b2, float* __restrict__ out, int E) {
    __shared__ SMem sm;

    const int t  = threadIdx.x;
    const int wg = blockIdx.x;

    // ---- gather x[row] ++ x[col] for 64 edges, split bf16 hi/lo ----
    {
        int m = t >> 2, q = t & 3;                        // 4 threads per edge row
        long long e     = (long long)wg * MWG + m;
        long long esafe = (e < E) ? e : (long long)(E - 1);
        long long r = ei[esafe];
        long long c = ei[(long long)E + esafe];
        const float* rowp = x + r * IN_CH;
        const float* colp = x + c * IN_CH;
        char* zhb = (char*)&sm.g.z[m][0][0];
        char* zlb = (char*)&sm.g.z[m][1][0];
        #pragma unroll
        for (int u = 0; u < 16; ++u) {
            int fi = u * 4 + q;                           // float4 index 0..63
            const float* src = (fi < 32) ? (rowp + fi * 4) : (colp + (fi - 32) * 4);
            v4f v = *(const v4f*)src;
            unsigned u0 = __builtin_bit_cast(unsigned, v[0]);
            unsigned u1 = __builtin_bit_cast(unsigned, v[1]);
            unsigned u2 = __builtin_bit_cast(unsigned, v[2]);
            unsigned u3 = __builtin_bit_cast(unsigned, v[3]);
            uint2 hp;                                     // 4 packed bf16 hi
            hp.x = (u0 >> 16) | (u1 & 0xffff0000u);
            hp.y = (u2 >> 16) | (u3 & 0xffff0000u);
            float l0 = v[0] - __builtin_bit_cast(float, u0 & 0xffff0000u);
            float l1 = v[1] - __builtin_bit_cast(float, u1 & 0xffff0000u);
            float l2 = v[2] - __builtin_bit_cast(float, u2 & 0xffff0000u);
            float l3 = v[3] - __builtin_bit_cast(float, u3 & 0xffff0000u);
            uint2 lp;                                     // 4 packed bf16 lo
            lp.x = (__builtin_bit_cast(unsigned, l0) >> 16) |
                   (__builtin_bit_cast(unsigned, l1) & 0xffff0000u);
            lp.y = (__builtin_bit_cast(unsigned, l2) >> 16) |
                   (__builtin_bit_cast(unsigned, l3) & 0xffff0000u);
            *(uint2*)(zhb + fi * 8) = hp;                 // 8B DS stores
            *(uint2*)(zlb + fi * 8) = lp;
        }
    }
    __syncthreads();

    // ---- layer 1: bf16 hi/lo split WMMA (3 per product tile) ----
    const int wave = t >> 5;
    const int lane = t & 31;
    const int kh   = lane >> 4;
    const int l16  = lane & 15;
    const int nt0  = wave;
    const int nt1  = wave + 8;
    const __bf16* Bh = wsB;
    const __bf16* Bl = wsB + WFRAG;

    // single per-lane A base; all fragment offsets are compile-time consts
    const v8bf* zbase = &sm.g.z[l16][0][kh];              // byte: l16*1056 + kh*16

    v8f acc[MTILES][2];
    #pragma unroll
    for (int mt = 0; mt < MTILES; ++mt) { acc[mt][0] = (v8f){}; acc[mt][1] = (v8f){}; }

    #pragma unroll
    for (int ks = 0; ks < KSTEPS; ++ks) {
        // B fragments for this K-step (L2-resident, reused across 4 M-tiles)
        const __bf16* p0 = Bh + (((nt0 * KSTEPS + ks) * 32 + lane) << 4);
        const __bf16* p1 = Bl + (((nt0 * KSTEPS + ks) * 32 + lane) << 4);
        const __bf16* p2 = Bh + (((nt1 * KSTEPS + ks) * 32 + lane) << 4);
        const __bf16* p3 = Bl + (((nt1 * KSTEPS + ks) * 32 + lane) << 4);
        v16bf bh0 = cat8(*(const v8bf*)p0, *(const v8bf*)(p0 + 8));
        v16bf bl0 = cat8(*(const v8bf*)p1, *(const v8bf*)(p1 + 8));
        v16bf bh1 = cat8(*(const v8bf*)p2, *(const v8bf*)(p2 + 8));
        v16bf bl1 = cat8(*(const v8bf*)p3, *(const v8bf*)(p3 + 8));

        #pragma unroll
        for (int mt = 0; mt < MTILES; ++mt) {
            const int o = mt * (16 * 2 * ZROW) + ks * 4;  // v8bf units, const per iter
            v16bf ah = cat8(zbase[o],        zbase[o + 2]);
            v16bf al = cat8(zbase[o + ZROW], zbase[o + ZROW + 2]);

            acc[mt][0] = __builtin_amdgcn_wmma_f32_16x16x32_bf16(false, ah, false, bh0, (short)0, acc[mt][0], false, false);
            acc[mt][0] = __builtin_amdgcn_wmma_f32_16x16x32_bf16(false, ah, false, bl0, (short)0, acc[mt][0], false, false);
            acc[mt][0] = __builtin_amdgcn_wmma_f32_16x16x32_bf16(false, al, false, bh0, (short)0, acc[mt][0], false, false);
            acc[mt][1] = __builtin_amdgcn_wmma_f32_16x16x32_bf16(false, ah, false, bh1, (short)0, acc[mt][1], false, false);
            acc[mt][1] = __builtin_amdgcn_wmma_f32_16x16x32_bf16(false, ah, false, bl1, (short)0, acc[mt][1], false, false);
            acc[mt][1] = __builtin_amdgcn_wmma_f32_16x16x32_bf16(false, al, false, bh1, (short)0, acc[mt][1], false, false);
        }
    }
    __syncthreads();   // all z reads done; LDS reused as h/red

    // ---- bias + ReLU, scatter D tiles to LDS ----
    {
        float bias0 = b1[nt0 * 16 + l16];
        float bias1 = b1[nt1 * 16 + l16];
        #pragma unroll
        for (int mt = 0; mt < MTILES; ++mt) {
            #pragma unroll
            for (int j = 0; j < 8; ++j) {
                int m = mt * 16 + kh * 8 + j;
                sm.r.h[m][nt0 * 16 + l16] = fmaxf(acc[mt][0][j] + bias0, 0.0f);
                sm.r.h[m][nt1 * 16 + l16] = fmaxf(acc[mt][1][j] + bias1, 0.0f);
            }
        }
    }
    __syncthreads();

    // ---- layer 2: out[m] = b2 + sum_n h[m][n] * W2[n] ----
    {
        int m = t & 63, seg = t >> 6;                     // conflict-free: stride 257
        float s = 0.0f;
        #pragma unroll
        for (int j = 0; j < 64; ++j)
            s += sm.r.h[m][seg * 64 + j] * W2[seg * 64 + j];
        sm.r.red[m][seg] = s;
    }
    __syncthreads();
    if (t < MWG) {
        long long e = (long long)wg * MWG + t;
        if (e < E) {
            out[e] = b2[0] + sm.r.red[t][0] + sm.r.red[t][1]
                           + sm.r.red[t][2] + sm.r.red[t][3];
        }
    }
}

// ---------------------------------------------------------------------------
extern "C" void kernel_launch(void* const* d_in, const int* in_sizes, int n_in,
                              void* d_out, int out_size, void* d_ws, size_t ws_size,
                              hipStream_t stream) {
    const long long* ei = (const long long*)d_in[0];   // int64 [2][E]
    const float* x  = (const float*)d_in[1];
    const float* W1 = (const float*)d_in[2];
    const float* b1 = (const float*)d_in[3];
    const float* W2 = (const float*)d_in[4];
    const float* b2 = (const float*)d_in[5];
    float* out = (float*)d_out;
    int E = in_sizes[0] / 2;

    __bf16* wsB = (__bf16*)d_ws;   // needs 2*WFRAG*2 = 256 KB of scratch

    repack_w1<<<256, 256, 0, stream>>>(W1, wsB);
    int blocks = (E + MWG - 1) / MWG;
    edge_mlp<<<blocks, 256, 0, stream>>>(ei, x, wsB, b1, W2, b2, out, E);
}
